// QuantizedMultimodalCrossAttentionKernel_89223650607644
// MI455X (gfx1250) — matrix operand order repaired and checked by reading, not compiled, not run on hardware
//
#include <hip/hip_runtime.h>
#include <hip/hip_bf16.h>

// ---------------- problem constants ----------------
#define Dm   2048
#define Hh   16
#define HDd  128
#define Bb   4
#define STt  512
#define SIi  576
#define SKV  1088          // ST + SI
#define SCALING 0.08838834764831845f   // 128^-0.5
#define LN_EPS 1e-5f

typedef __attribute__((ext_vector_type(16))) _Float16 v16h;
typedef __attribute__((ext_vector_type(8)))  float    v8f;

union FragH {
    v16h v;
    unsigned int u[8];
    _Float16 h[16];
};
union PairH {                 // two f16 packed in one dword (ds_store_b32)
    unsigned int u;
    _Float16 h[2];
};
union OctH {                  // eight f16 packed in 16B (ds_store_b128)
    uint4 q;
    _Float16 h[8];
};

// ------------------------------------------------------------------
// Fragment loaders (wave32, CDNA5 16-bit WMMA layouts, CDNA5 ISA)
// A 16x32 f16: lanes 0-15 -> M, V0..3 = K{0..7}+8*hi paired, V4..7 = K{16..23}+8*hi
// B 32x16 f16: lanes 0-15 -> N, elems 0..7 = K{0..7}+8*hi, elems 8..15 = K{16..23}+8*hi
// ------------------------------------------------------------------
__device__ __forceinline__ v16h load_afrag(const _Float16* rowBase, int hi) {
    FragH a;
#pragma unroll
    for (int v = 0; v < 8; ++v) {
        int k = (v < 4) ? (2 * v + 8 * hi) : (2 * v + 8 + 8 * hi);
        a.u[v] = *reinterpret_cast<const unsigned int*>(&rowBase[k]);
    }
    return a.v;
}

__device__ __forceinline__ v16h load_bfrag(const _Float16* colBase, int ld, int hi) {
    FragH b;
#pragma unroll
    for (int s = 0; s < 16; ++s) {
        int k = (s < 8) ? (s + 8 * hi) : (s + 8 + 8 * hi);
        b.h[s] = colBase[k * ld];
    }
    return b.v;
}

// ------------------------------------------------------------------
// Kernel 1: LayerNorm -> f16 activations
// ------------------------------------------------------------------
__global__ __launch_bounds__(256)
void layernorm_f16(const float* __restrict__ x, const float* __restrict__ g,
                   const float* __restrict__ bta, _Float16* __restrict__ out) {
    __shared__ float red[256];
    const int tid = threadIdx.x;
    const float* row = x + (size_t)blockIdx.x * Dm;
    float v[8];
    float s = 0.f;
#pragma unroll
    for (int i = 0; i < 8; ++i) { v[i] = row[tid + i * 256]; s += v[i]; }
    red[tid] = s; __syncthreads();
    for (int st = 128; st > 0; st >>= 1) {
        if (tid < st) red[tid] += red[tid + st];
        __syncthreads();
    }
    const float mean = red[0] * (1.0f / Dm);
    __syncthreads();
    float s2 = 0.f;
#pragma unroll
    for (int i = 0; i < 8; ++i) { float d = v[i] - mean; s2 += d * d; }
    red[tid] = s2; __syncthreads();
    for (int st = 128; st > 0; st >>= 1) {
        if (tid < st) red[tid] += red[tid + st];
        __syncthreads();
    }
    const float rstd = rsqrtf(red[0] * (1.0f / Dm) + LN_EPS);
    _Float16* orow = out + (size_t)blockIdx.x * Dm;
#pragma unroll
    for (int i = 0; i < 8; ++i) {
        int c = tid + i * 256;
        orow[c] = (_Float16)((v[i] - mean) * rstd * g[c] + bta[c]);
    }
}

// ------------------------------------------------------------------
// Kernel 2: quantized-dequant GEMM  out[m][n] = sum_k A[m][k]*W[n][k]*scale[n]+bias[n]
// A: f16 [M][D] ; W: int8-in-int32 [N=D][K=D] transposed+converted on LDS staging.
// Block tile 128x128x32, 8 waves of 32x64 (2x4 WMMA accums).
// mode 0: store f16 to [B,H,Sout,HD] head layout (kvOff for concat, qscale for Q)
// mode 1: store fp32 to [M][D] (final projection -> d_out)
// ------------------------------------------------------------------
#define GBM 128
#define GBN 128
#define GBK 32
#define GLDA 40    // GBK + 8
#define GLDB 136   // GBN + 8

__global__ __launch_bounds__(256)
void gemm_qlinear(const _Float16* __restrict__ A, const int* __restrict__ W,
                  const float* __restrict__ scale, const float* __restrict__ bias,
                  _Float16* __restrict__ outH, float* __restrict__ outF,
                  int Sq, int Sout, int kvOff, float qscale, int mode) {
    __shared__ __align__(16) _Float16 As[GBM * GLDA];
    __shared__ __align__(16) _Float16 Bs[GBK * GLDB];

    const int tid    = threadIdx.x;
    const int mBlock = blockIdx.y * GBM;
    const int nBlock = blockIdx.x * GBN;
    const int lane = tid & 31, lo = lane & 15, hi = lane >> 4;
    const int wv = tid >> 5;
    const int wm = (wv >> 1) * 32;   // 0,32,64,96
    const int wn = (wv & 1) * 64;    // 0,64

    v8f acc[2][4] = {};

    // B-staging thread mapping: each thread owns two adjacent n columns x 8 k rows
    const int bn0 = (tid & 63) * 2;        // even column 0..126
    const int bkc = (tid >> 6) * 8;        // k chunk: 0,8,16,24

    for (int k0 = 0; k0 < Dm; k0 += GBK) {
        // -- stage A tile: 128 rows x 32 halfs (b128 copies)
#pragma unroll
        for (int it = 0; it < 2; ++it) {
            int seg = tid + it * 256;          // 0..511
            int r = seg >> 2, sg = seg & 3;
            *reinterpret_cast<uint4*>(&As[r * GLDA + sg * 8]) =
                *reinterpret_cast<const uint4*>(&A[(size_t)(mBlock + r) * Dm + k0 + sg * 8]);
        }
        // -- stage B tile transposed + int->f16 convert: Bs[k][n] = W[n][k]
        //    packed pairs along n -> ds_store_b32
        {
            const int* w0 = &W[(size_t)(nBlock + bn0) * Dm + k0 + bkc];
            const int* w1 = w0 + Dm;
            int4 a0 = *reinterpret_cast<const int4*>(&w0[0]);
            int4 a1 = *reinterpret_cast<const int4*>(&w0[4]);
            int4 b0 = *reinterpret_cast<const int4*>(&w1[0]);
            int4 b1 = *reinterpret_cast<const int4*>(&w1[4]);
            PairH p;
            p.h[0] = (_Float16)a0.x; p.h[1] = (_Float16)b0.x;
            *reinterpret_cast<unsigned*>(&Bs[(bkc + 0) * GLDB + bn0]) = p.u;
            p.h[0] = (_Float16)a0.y; p.h[1] = (_Float16)b0.y;
            *reinterpret_cast<unsigned*>(&Bs[(bkc + 1) * GLDB + bn0]) = p.u;
            p.h[0] = (_Float16)a0.z; p.h[1] = (_Float16)b0.z;
            *reinterpret_cast<unsigned*>(&Bs[(bkc + 2) * GLDB + bn0]) = p.u;
            p.h[0] = (_Float16)a0.w; p.h[1] = (_Float16)b0.w;
            *reinterpret_cast<unsigned*>(&Bs[(bkc + 3) * GLDB + bn0]) = p.u;
            p.h[0] = (_Float16)a1.x; p.h[1] = (_Float16)b1.x;
            *reinterpret_cast<unsigned*>(&Bs[(bkc + 4) * GLDB + bn0]) = p.u;
            p.h[0] = (_Float16)a1.y; p.h[1] = (_Float16)b1.y;
            *reinterpret_cast<unsigned*>(&Bs[(bkc + 5) * GLDB + bn0]) = p.u;
            p.h[0] = (_Float16)a1.z; p.h[1] = (_Float16)b1.z;
            *reinterpret_cast<unsigned*>(&Bs[(bkc + 6) * GLDB + bn0]) = p.u;
            p.h[0] = (_Float16)a1.w; p.h[1] = (_Float16)b1.w;
            *reinterpret_cast<unsigned*>(&Bs[(bkc + 7) * GLDB + bn0]) = p.u;
        }
        if (k0 + GBK < Dm) {   // gfx1250 global_prefetch_b8 of next tiles
            __builtin_prefetch(&A[(size_t)(mBlock + (tid & 127)) * Dm + k0 + GBK], 0, 1);
            __builtin_prefetch(&W[(size_t)(nBlock + bn0) * Dm + k0 + GBK], 0, 1);
        }
        __syncthreads();

        v16h af[2], bf[4];
#pragma unroll
        for (int mt = 0; mt < 2; ++mt)
            af[mt] = load_afrag(&As[(wm + mt * 16 + lo) * GLDA], hi);
#pragma unroll
        for (int nt = 0; nt < 4; ++nt)
            bf[nt] = load_bfrag(&Bs[wn + nt * 16 + lo], GLDB, hi);
#pragma unroll
        for (int mt = 0; mt < 2; ++mt)
#pragma unroll
            for (int nt = 0; nt < 4; ++nt)
                acc[mt][nt] = __builtin_amdgcn_wmma_f32_16x16x32_f16(
                    false, af[mt], false, bf[nt], (short)0, acc[mt][nt], false, false);
        __syncthreads();
    }

    // -- epilogue
#pragma unroll
    for (int nt = 0; nt < 4; ++nt) {
        const int n = nBlock + wn + nt * 16 + lo;
        const float sc = scale[n], bi = bias[n];
        const int h = n >> 7, hd = n & (HDd - 1);
#pragma unroll
        for (int mt = 0; mt < 2; ++mt) {
#pragma unroll
            for (int j = 0; j < 8; ++j) {
                const int m = mBlock + wm + mt * 16 + j + 8 * hi;
                float val = (acc[mt][nt][j] * sc + bi) * qscale;
                if (mode == 0) {
                    const int b = m / Sq, s = m - b * Sq;
                    outH[(((size_t)(b * Hh + h)) * Sout + kvOff + s) * HDd + hd] = (_Float16)val;
                } else {
                    outF[(size_t)m * Dm + n] = val;
                }
            }
        }
    }
}

// ------------------------------------------------------------------
// Kernel 3: attention scores  attn[bh][m][n] = sum_hd q[bh][m][hd]*k[bh][n][hd]
// Block: 64x64 tile, K = HD = 128 staged once (K transposed into B layout).
// ------------------------------------------------------------------
__global__ __launch_bounds__(256)
void attn_scores(const _Float16* __restrict__ Q, const _Float16* __restrict__ Kc,
                 float* __restrict__ attn, int Sq) {
    __shared__ __align__(16) _Float16 Qs[64 * 136];
    __shared__ __align__(16) _Float16 Ks[128 * 72];

    const int tid = threadIdx.x;
    const int nBase = blockIdx.x * 64;
    const int qBase = blockIdx.y * 64;
    const int bh = blockIdx.z;
    const int lane = tid & 31, lo = lane & 15, hi = lane >> 4;
    const int wv = tid >> 5;
    const int wm = (wv >> 1) * 16;   // 0..48
    const int wn = (wv & 1) * 32;    // 0,32

    // stage Q tile [64][128]
#pragma unroll
    for (int it = 0; it < 4; ++it) {
        int seg = tid + it * 256;            // 0..1023
        int r = seg >> 4, sg = seg & 15;
        *reinterpret_cast<uint4*>(&Qs[r * 136 + sg * 8]) =
            *reinterpret_cast<const uint4*>(&Q[((size_t)bh * Sq + qBase + r) * HDd + sg * 8]);
    }
    // stage K tile transposed: Ks[hd][n] = Kc[bh][nBase+n][hd]
    // each thread: two adjacent n columns x 16 hd rows, packed b32 stores
    {
        const int n0 = (tid & 31) * 2;       // 0..62
        const int hc = (tid >> 5) * 16;      // 0..112
        const _Float16* s0 = &Kc[((size_t)bh * SKV + nBase + n0) * HDd + hc];
        const _Float16* s1 = s0 + HDd;
#pragma unroll
        for (int j = 0; j < 16; ++j) {
            PairH p;
            p.h[0] = s0[j];
            p.h[1] = s1[j];
            *reinterpret_cast<unsigned*>(&Ks[(hc + j) * 72 + n0]) = p.u;
        }
    }
    __syncthreads();

    v8f acc[2] = {};
#pragma unroll
    for (int kk = 0; kk < 4; ++kk) {
        const int k0 = kk * 32;
        v16h af = load_afrag(&Qs[(wm + lo) * 136 + k0], hi);
#pragma unroll
        for (int nt = 0; nt < 2; ++nt) {
            v16h bf = load_bfrag(&Ks[k0 * 72 + wn + nt * 16 + lo], 72, hi);
            acc[nt] = __builtin_amdgcn_wmma_f32_16x16x32_f16(
                false, af, false, bf, (short)0, acc[nt], false, false);
        }
    }

#pragma unroll
    for (int nt = 0; nt < 2; ++nt) {
        const int n = nBase + wn + nt * 16 + lo;
#pragma unroll
        for (int j = 0; j < 8; ++j) {
            const int m = qBase + wm + j + 8 * hi;
            attn[((size_t)bh * Sq + m) * SKV + n] = acc[nt][j];
        }
    }
}

// ------------------------------------------------------------------
// Kernel 4: in-place row softmax (rows of length SKV)
// ------------------------------------------------------------------
__global__ __launch_bounds__(256)
void softmax_rows(float* __restrict__ p) {
    __shared__ float red[256];
    const int tid = threadIdx.x;
    float* row = p + (size_t)blockIdx.x * SKV;
    float mx = -1e30f;
    for (int i = tid; i < SKV; i += 256) mx = fmaxf(mx, row[i]);
    red[tid] = mx; __syncthreads();
    for (int st = 128; st > 0; st >>= 1) {
        if (tid < st) red[tid] = fmaxf(red[tid], red[tid + st]);
        __syncthreads();
    }
    mx = red[0]; __syncthreads();
    float sum = 0.f;
    for (int i = tid; i < SKV; i += 256) {
        float e = __expf(row[i] - mx);
        row[i] = e;
        sum += e;
    }
    red[tid] = sum; __syncthreads();
    for (int st = 128; st > 0; st >>= 1) {
        if (tid < st) red[tid] += red[tid + st];
        __syncthreads();
    }
    const float inv = 1.0f / red[0];
    for (int i = tid; i < SKV; i += 256) row[i] *= inv;
}

// ------------------------------------------------------------------
// Kernel 5: out[bh][m][hd] = sum_k attn[bh][m][k]*v[bh][k][hd]
// Block 64x128 tile, K = SKV in 34 steps of 32. Output to [B,S,D] token layout.
// ------------------------------------------------------------------
__global__ __launch_bounds__(256)
void attn_av(const float* __restrict__ attnP, const _Float16* __restrict__ Vc,
             _Float16* __restrict__ ao, int Sq) {
    __shared__ __align__(16) _Float16 As[64 * 40];
    __shared__ __align__(16) _Float16 Bs[32 * 136];

    const int tid = threadIdx.x;
    const int qBase = blockIdx.x * 64;
    const int bh = blockIdx.y;
    const int lane = tid & 31, lo = lane & 15, hi = lane >> 4;
    const int wv = tid >> 5;
    const int wm = (wv >> 1) * 16;   // 0..48
    const int wn = (wv & 1) * 64;    // 0,64

    v8f acc[4] = {};

    const int ar = tid >> 2, akc = (tid & 3) * 8;   // A staging map
    const int bk = tid >> 3, bsg = tid & 7;         // B staging map

    for (int k0 = 0; k0 < SKV; k0 += 32) {
        // stage attn chunk fp32 -> f16, packed into one ds_store_b128
        {
            const float* src = &attnP[((size_t)bh * Sq + qBase + ar) * SKV + k0 + akc];
            float4 f0 = *reinterpret_cast<const float4*>(src);
            float4 f1 = *reinterpret_cast<const float4*>(src + 4);
            OctH pk;
            pk.h[0] = (_Float16)f0.x; pk.h[1] = (_Float16)f0.y;
            pk.h[2] = (_Float16)f0.z; pk.h[3] = (_Float16)f0.w;
            pk.h[4] = (_Float16)f1.x; pk.h[5] = (_Float16)f1.y;
            pk.h[6] = (_Float16)f1.z; pk.h[7] = (_Float16)f1.w;
            *reinterpret_cast<uint4*>(&As[ar * 40 + akc]) = pk.q;
        }
        // stage V chunk (already B-layout friendly: rows = k, contiguous hd)
        *reinterpret_cast<uint4*>(&Bs[bk * 136 + bsg * 16]) =
            *reinterpret_cast<const uint4*>(&Vc[((size_t)bh * SKV + k0 + bk) * HDd + bsg * 16]);
        __syncthreads();

        v16h af = load_afrag(&As[(wm + lo) * 40], hi);
#pragma unroll
        for (int nt = 0; nt < 4; ++nt) {
            v16h bf = load_bfrag(&Bs[wn + nt * 16 + lo], 136, hi);
            acc[nt] = __builtin_amdgcn_wmma_f32_16x16x32_f16(
                false, af, false, bf, (short)0, acc[nt], false, false);
        }
        __syncthreads();
    }

    const int b = bh >> 4, h = bh & 15;
#pragma unroll
    for (int nt = 0; nt < 4; ++nt) {
        const int n = wn + nt * 16 + lo;     // hd index
#pragma unroll
        for (int j = 0; j < 8; ++j) {
            const int m = qBase + wm + j + 8 * hi;
            ao[((size_t)(b * Sq + m)) * Dm + h * HDd + n] = (_Float16)acc[nt][j];
        }
    }
}

// ------------------------------------------------------------------
// Host launcher
// ------------------------------------------------------------------
extern "C" void kernel_launch(void* const* d_in, const int* in_sizes, int n_in,
                              void* d_out, int out_size, void* d_ws, size_t ws_size,
                              hipStream_t stream) {
    const float* text    = (const float*)d_in[0];
    const float* image   = (const float*)d_in[1];
    const float* ln_g_t  = (const float*)d_in[2];
    const float* ln_b_t  = (const float*)d_in[3];
    const float* sq_t    = (const float*)d_in[4];
    const float* bq_t    = (const float*)d_in[5];
    const int*   wq_t    = (const int*)  d_in[6];
    const float* sk_t    = (const float*)d_in[7];
    const float* bk_t    = (const float*)d_in[8];
    const int*   wk_t    = (const int*)  d_in[9];
    const float* sv_t    = (const float*)d_in[10];
    const float* bv_t    = (const float*)d_in[11];
    const int*   wv_t    = (const int*)  d_in[12];
    const float* ln_g_i  = (const float*)d_in[13];
    const float* ln_b_i  = (const float*)d_in[14];
    const float* sq_i    = (const float*)d_in[15];
    const float* bq_i    = (const float*)d_in[16];
    const int*   wq_i    = (const int*)  d_in[17];
    const float* sk_i    = (const float*)d_in[18];
    const float* bk_i    = (const float*)d_in[19];
    const int*   wk_i    = (const int*)  d_in[20];
    const float* sv_i    = (const float*)d_in[21];
    const float* bv_i    = (const float*)d_in[22];
    const int*   wv_i    = (const int*)  d_in[23];
    const float* so      = (const float*)d_in[24];
    const float* bo      = (const float*)d_in[25];
    const int*   wo      = (const int*)  d_in[26];

    // workspace layout (bytes)
    char* base = (char*)d_ws;
    _Float16* nt  = (_Float16*)(base);                       // [B*ST, D]
    _Float16* ni  = (_Float16*)(base + 8388608);             // [B*SI, D]
    _Float16* qT  = (_Float16*)(base + 17825792);            // [B,H,ST,HD]
    _Float16* qI  = (_Float16*)(base + 26214400);            // [B,H,SI,HD]
    _Float16* kc  = (_Float16*)(base + 35651584);            // [B,H,SKV,HD]
    _Float16* vc  = (_Float16*)(base + 53477376);            // [B,H,SKV,HD]
    _Float16* aoT = (_Float16*)(base + 71303168);            // [B*ST, D]
    _Float16* aoI = (_Float16*)(base + 79691776);            // [B*SI, D]

    // output layout (floats): out_text, out_image, attn_text, attn_image
    float* outT  = (float*)d_out;
    float* outI  = outT + (size_t)Bb * STt * Dm;             // + 4194304
    float* attnT = outI + (size_t)Bb * SIi * Dm;             // + 4718592
    float* attnI = attnT + (size_t)Bb * Hh * STt * SKV;      // + 35651584

    // 1) layernorm
    layernorm_f16<<<Bb * STt, 256, 0, stream>>>(text,  ln_g_t, ln_b_t, nt);
    layernorm_f16<<<Bb * SIi, 256, 0, stream>>>(image, ln_g_i, ln_b_i, ni);

    // 2) Q/K/V projections (WMMA int8-dequant GEMM, head transpose fused)
    dim3 gT(Dm / GBN, (Bb * STt) / GBM);   // (16, 16)
    dim3 gI(Dm / GBN, (Bb * SIi) / GBM);   // (16, 18)
    gemm_qlinear<<<gT, 256, 0, stream>>>(nt, wq_t, sq_t, bq_t, qT, nullptr, STt, STt, 0,   SCALING, 0);
    gemm_qlinear<<<gT, 256, 0, stream>>>(nt, wk_t, sk_t, bk_t, kc, nullptr, STt, SKV, 0,   1.0f,    0);
    gemm_qlinear<<<gT, 256, 0, stream>>>(nt, wv_t, sv_t, bv_t, vc, nullptr, STt, SKV, 0,   1.0f,    0);
    gemm_qlinear<<<gI, 256, 0, stream>>>(ni, wq_i, sq_i, bq_i, qI, nullptr, SIi, SIi, 0,   SCALING, 0);
    gemm_qlinear<<<gI, 256, 0, stream>>>(ni, wk_i, sk_i, bk_i, kc, nullptr, SIi, SKV, STt, 1.0f,    0);
    gemm_qlinear<<<gI, 256, 0, stream>>>(ni, wv_i, sv_i, bv_i, vc, nullptr, SIi, SKV, STt, 1.0f,    0);

    // 3) attention scores (WMMA) -> fp32 attn regions of d_out
    attn_scores<<<dim3(SKV / 64, STt / 64, Bb * Hh), 256, 0, stream>>>(qT, kc, attnT, STt);
    attn_scores<<<dim3(SKV / 64, SIi / 64, Bb * Hh), 256, 0, stream>>>(qI, kc, attnI, SIi);

    // 4) softmax in place
    softmax_rows<<<Bb * Hh * STt, 256, 0, stream>>>(attnT);
    softmax_rows<<<Bb * Hh * SIi, 256, 0, stream>>>(attnI);

    // 5) attn @ V (WMMA), output back to token-major [B,S,D] f16
    attn_av<<<dim3(STt / 64, Bb * Hh), 256, 0, stream>>>(attnT, vc, aoT, STt);
    attn_av<<<dim3(SIi / 64, Bb * Hh), 256, 0, stream>>>(attnI, vc, aoI, SIi);

    // 6) output projection -> fp32 d_out
    gemm_qlinear<<<gT, 256, 0, stream>>>(aoT, wo, so, bo, nullptr, outT, STt, 0, 0, 1.0f, 1);
    gemm_qlinear<<<gI, 256, 0, stream>>>(aoI, wo, so, bo, nullptr, outI, SIi, 0, 0, 1.0f, 1);
}
